// QuantMLP_19799799235060
// MI455X (gfx1250) — 8-bit matrix operand order repaired and checked by reading, compiled, not run on hardware
//
#include <hip/hip_runtime.h>
#include <stdint.h>


typedef __attribute__((ext_vector_type(8))) int v8i;

#define BATCH 4096
#define D_IN  2048
#define H_DIM 4096
#define D_OUT 2048


__global__ __launch_bounds__(256) void quant_x_kernel(
    const float* __restrict__ x, const float* __restrict__ amax_p,
    int8_t* __restrict__ xq, int n4) {
  int i = blockIdx.x * blockDim.x + threadIdx.x;
  if (i >= n4) return;
  const float s = 127.0f / *amax_p;
  const float4 v = reinterpret_cast<const float4*>(x)[i];
  char4 q;
  q.x = (signed char)fminf(fmaxf(rintf(v.x * s), -127.0f), 127.0f);
  q.y = (signed char)fminf(fmaxf(rintf(v.y * s), -127.0f), 127.0f);
  q.z = (signed char)fminf(fmaxf(rintf(v.z * s), -127.0f), 127.0f);
  q.w = (signed char)fminf(fmaxf(rintf(v.w * s), -127.0f), 127.0f);
  reinterpret_cast<char4*>(xq)[i] = q;
}

__global__ __launch_bounds__(256) void pack_w_kernel(
    const int* __restrict__ w, int8_t* __restrict__ wq, int n4) {
  int i = blockIdx.x * blockDim.x + threadIdx.x;
  if (i >= n4) return;
  const int4 v = reinterpret_cast<const int4*>(w)[i];
  char4 q;
  q.x = (signed char)v.x;
  q.y = (signed char)v.y;
  q.z = (signed char)v.z;
  q.w = (signed char)v.w;
  reinterpret_cast<char4*>(wq)[i] = q;
}


__device__ __forceinline__ v8i load_a_frag(const int8_t* row_base, int half) {
  v8i f;
#pragma unroll
  for (int v = 0; v < 8; ++v) {
    const int kb = ((v >> 1) * 16) + half * 8 + ((v & 1) * 4);
    f[v] = *reinterpret_cast<const int*>(row_base + kb);
  }
  return f;
}

__device__ __forceinline__ v8i load_b_frag(const int8_t* row_base, int half) {
  v8i f;
#pragma unroll
  for (int v = 0; v < 8; ++v) {
    const int kb = ((v >> 1) * 16) + half * 8 + ((v & 1) * 4);
    f[v] = *reinterpret_cast<const int*>(row_base + kb);
  }
  return f;
}

template <bool FUSE_RELU_QUANT>
__global__ __launch_bounds__(256) void gemm_i8_kernel(
    const int8_t* __restrict__ A, const int8_t* __restrict__ W,
    const int* __restrict__ bias,
    const float* __restrict__ a_in_p, const float* __restrict__ a_w_p,
    const float* __restrict__ a_b_p, const float* __restrict__ a_next_p,
    int K, int N, int8_t* __restrict__ outq, float* __restrict__ outf) {
  const int lane = threadIdx.x & 31;
  const int wave = threadIdx.x >> 5;
  const int half = lane >> 4;
  const int ln16 = lane & 15;

  const int m0 = blockIdx.y * 256 + (wave >> 1) * 64;
  const int n0 = blockIdx.x * 128 + (wave & 1) * 64;

  const float a_in = *a_in_p;
  const float a_w = *a_w_p;
  const float a_b = *a_b_p;
  const float acc_scale = a_w * a_in * (1.0f / 16129.0f);
  const float bias_scale = a_b * (1.0f / 127.0f);
  float qscale = 0.0f;
  if (FUSE_RELU_QUANT) qscale = 127.0f / *a_next_p;

  const int8_t* arow[4];
  const int8_t* brow[4];
#pragma unroll
  for (int r = 0; r < 4; ++r)
    arow[r] = A + (size_t)(m0 + r * 16 + ln16) * (size_t)K;
#pragma unroll
  for (int c = 0; c < 4; ++c)
    brow[c] = W + (size_t)(n0 + c * 16 + ln16) * (size_t)K;

  const v8i zero = {0, 0, 0, 0, 0, 0, 0, 0};
  v8i acc[4][4];
#pragma unroll
  for (int r = 0; r < 4; ++r)
#pragma unroll
    for (int c = 0; c < 4; ++c) acc[r][c] = zero;

  for (int k0 = 0; k0 < K; k0 += 64) {
    if (k0 + 64 < K) {
#pragma unroll
      for (int r = 0; r < 4; ++r) __builtin_prefetch(arow[r] + k0 + 64, 0, 1);
#pragma unroll
      for (int c = 0; c < 4; ++c) __builtin_prefetch(brow[c] + k0 + 64, 0, 1);
    }
    v8i afrag[4], bfrag[4];
#pragma unroll
    for (int r = 0; r < 4; ++r) afrag[r] = load_a_frag(arow[r] + k0, half);
#pragma unroll
    for (int c = 0; c < 4; ++c) bfrag[c] = load_b_frag(brow[c] + k0, half);
#pragma unroll
    for (int r = 0; r < 4; ++r)
#pragma unroll
      for (int c = 0; c < 4; ++c)
        acc[r][c] = __builtin_amdgcn_wmma_i32_16x16x64_iu8(
             true, afrag[r],  true, bfrag[c], acc[r][c],
             false,  false);
  }

#pragma unroll
  for (int c = 0; c < 4; ++c) {
    const int n = n0 + c * 16 + ln16;
    const float bterm = (float)bias[n] * bias_scale;
#pragma unroll
    for (int r = 0; r < 4; ++r) {
#pragma unroll
      for (int j = 0; j < 8; ++j) {
        const int row = m0 + r * 16 + half * 8 + j;
        float y = (float)acc[r][c][j] * acc_scale + bterm;
        if (FUSE_RELU_QUANT) {
          y = fmaxf(y, 0.0f);
          const float q = fminf(fmaxf(rintf(y * qscale), -127.0f), 127.0f);
          outq[(size_t)row * (size_t)N + n] = (int8_t)(signed char)q;
        } else {
          outf[(size_t)row * (size_t)N + n] = y;
        }
      }
    }
  }
}


extern "C" void kernel_launch(void* const* d_in, const int* in_sizes, int n_in,
                              void* d_out, int out_size, void* d_ws,
                              size_t ws_size, hipStream_t stream) {
  (void)in_sizes; (void)n_in; (void)out_size; (void)ws_size;

  const float* x = (const float*)d_in[0];
  const int* W0 = (const int*)d_in[1];
  const int* b0 = (const int*)d_in[2];
  const int* W2 = (const int*)d_in[3];
  const int* b2 = (const int*)d_in[4];
  const int* W4 = (const int*)d_in[5];
  const int* b4 = (const int*)d_in[6];
  const float* a0_in = (const float*)d_in[7];
  const float* a0_w = (const float*)d_in[8];
  const float* a0_b = (const float*)d_in[9];
  const float* a2_in = (const float*)d_in[10];
  const float* a2_w = (const float*)d_in[11];
  const float* a2_b = (const float*)d_in[12];
  const float* a4_in = (const float*)d_in[13];
  const float* a4_w = (const float*)d_in[14];
  const float* a4_b = (const float*)d_in[15];
  float* out = (float*)d_out;

  int8_t* ws = (int8_t*)d_ws;
  const size_t szX0 = (size_t)BATCH * D_IN;
  const size_t szH = (size_t)BATCH * H_DIM;
  const size_t szW0 = (size_t)H_DIM * D_IN;
  const size_t szW2 = (size_t)H_DIM * H_DIM;
  const size_t szW4 = (size_t)D_OUT * H_DIM;
  int8_t* xq0 = ws;
  int8_t* xq1 = xq0 + szX0;
  int8_t* xq2 = xq1 + szH;
  int8_t* w0q = xq2 + szH;
  int8_t* w2q = w0q + szW0;
  int8_t* w4q = w2q + szW2;

  {
    const int n4 = (int)(szX0 / 4);
    quant_x_kernel<<<n4 / 256, 256, 0, stream>>>(x, a0_in, xq0, n4);
  }
  {
    int n4 = (int)(szW0 / 4);
    pack_w_kernel<<<n4 / 256, 256, 0, stream>>>(W0, w0q, n4);
    n4 = (int)(szW2 / 4);
    pack_w_kernel<<<n4 / 256, 256, 0, stream>>>(W2, w2q, n4);
    n4 = (int)(szW4 / 4);
    pack_w_kernel<<<n4 / 256, 256, 0, stream>>>(W4, w4q, n4);
  }
  {
    dim3 grid(H_DIM / 128, BATCH / 256);
    gemm_i8_kernel<true><<<grid, 256, 0, stream>>>(
        xq0, w0q, b0, a0_in, a0_w, a0_b, a2_in, D_IN, H_DIM, xq1, nullptr);
  }
  {
    dim3 grid(H_DIM / 128, BATCH / 256);
    gemm_i8_kernel<true><<<grid, 256, 0, stream>>>(
        xq1, w2q, b2, a2_in, a2_w, a2_b, a4_in, H_DIM, H_DIM, xq2, nullptr);
  }
  {
    dim3 grid(D_OUT / 128, BATCH / 256);
    gemm_i8_kernel<false><<<grid, 256, 0, stream>>>(
        xq2, w4q, b4, a4_in, a4_w, a4_b, a4_in, H_DIM, D_OUT, nullptr, out);
  }
}
